// TransformerBlockQuantum_65481071398583
// MI455X (gfx1250) — matrix operand order repaired and checked by reading, not compiled, not run on hardware
//
#include <hip/hip_runtime.h>
#include <hip/hip_bf16.h>
#include <math.h>

// ---------------- problem constants ----------------
#define B_   8
#define S_   2048
#define E_   768
#define FF_  3072
#define NQ_  128
#define ROWS_ (B_ * S_)          // 16384
#define EPS_ 1e-5f
#define SCALE_ 0.3535533905932738f   // 1/sqrt(8)

typedef __attribute__((ext_vector_type(16))) __bf16 bf16x16;
typedef __attribute__((ext_vector_type(8)))  float  f32x8;
typedef __attribute__((ext_vector_type(4)))  unsigned int u32x4;
typedef __attribute__((ext_vector_type(8)))  int i32x8;
typedef __attribute__((ext_vector_type(4)))  int i32x4;

// ---------------- WMMA wrapper --------------------
__device__ inline f32x8 wmma_bf16(bf16x16 a, bf16x16 b, f32x8 c) {
  // D = A(16x32 bf16) * B(32x16 bf16) + C(16x16 f32)
  return __builtin_amdgcn_wmma_f32_16x16x32_bf16(
      false, a, false, b, (short)0, c, false, false);
}

// ------------- fragment loaders (wave32) -----------
// A 16x32 bf16 as loaded here (contiguous k, not ISA order): lane L holds row M=L&15; VGPR v holds
// K = {2v, 2v+1} + 16*(L>>4).
__device__ inline bf16x16 ld_a_rm_f32(const float* p, int ld, int lane) {
  int row = lane & 15, kh = (lane >> 4) << 4;
  bf16x16 a;
#pragma unroll
  for (int v = 0; v < 8; ++v) {
    int k = 2 * v + kh;
    const float* q = p + (size_t)row * ld + k;
    a[2 * v]     = (__bf16)q[0];
    a[2 * v + 1] = (__bf16)q[1];
  }
  return a;
}
__device__ inline bf16x16 ld_a_rm_bf16(const __bf16* p, int ld, int lane) {
  int row = lane & 15, kh = (lane >> 4) << 4;
  bf16x16 a;
#pragma unroll
  for (int v = 0; v < 8; ++v) {
    int k = 2 * v + kh;
    const __bf16* q = p + (size_t)row * ld + k;
    a[2 * v]     = q[0];
    a[2 * v + 1] = q[1];
  }
  return a;
}
// B 32x16 bf16 as loaded here (same contiguous k order as A): lane L holds col N=L&15; K range [0,16) for lanes 0-15,
// [16,32) for lanes 16-31; VGPR v holds K = kb + {2v, 2v+1}.
// "nk" source: element (k,n) lives at p[n*ld + k]  (e.g. W[f,e] row-major)
__device__ inline bf16x16 ld_b_nk_f32(const float* p, int ld, int lane) {
  int n = lane & 15, kb = (lane >> 4) << 4;
  bf16x16 b;
#pragma unroll
  for (int v = 0; v < 8; ++v) {
    const float* q = p + (size_t)n * ld + kb + 2 * v;
    b[2 * v]     = (__bf16)q[0];
    b[2 * v + 1] = (__bf16)q[1];
  }
  return b;
}
__device__ inline bf16x16 ld_b_nk_bf16(const __bf16* p, int ld, int lane) {
  int n = lane & 15, kb = (lane >> 4) << 4;
  bf16x16 b;
#pragma unroll
  for (int v = 0; v < 8; ++v) {
    const __bf16* q = p + (size_t)n * ld + kb + 2 * v;
    b[2 * v]     = q[0];
    b[2 * v + 1] = q[1];
  }
  return b;
}
// "kn" source: element (k,n) lives at p[k*ld + n]  (e.g. V rows)
__device__ inline bf16x16 ld_b_kn_bf16(const __bf16* p, int ld, int lane) {
  int n = lane & 15, kb = (lane >> 4) << 4;
  bf16x16 b;
#pragma unroll
  for (int v = 0; v < 8; ++v) {
    b[2 * v]     = p[(size_t)(kb + 2 * v)     * ld + n];
    b[2 * v + 1] = p[(size_t)(kb + 2 * v + 1) * ld + n];
  }
  return b;
}

// ------------- wave reductions (wave32) ------------
__device__ inline float red_max16(float v) {   // within 16-lane half
#pragma unroll
  for (int m = 1; m <= 8; m <<= 1) v = fmaxf(v, __shfl_xor(v, m, 32));
  return v;
}
__device__ inline float red_sum16(float v) {
#pragma unroll
  for (int m = 1; m <= 8; m <<= 1) v += __shfl_xor(v, m, 32);
  return v;
}
__device__ inline float red_sum32(float v) {
#pragma unroll
  for (int m = 1; m <= 16; m <<= 1) v += __shfl_xor(v, m, 32);
  return v;
}

// ------------- Tensor Data Mover helper ------------
// 2D bf16 tile DMA: global (row-major, ld = E_ elements) -> LDS, contiguous.
// D# per CDNA5 ISA:
//   group0: count=1 | lds_addr | global_addr[56:0] | type=2
//   group1: data_size=2B, tensor_dim0=E_, tensor_dim1=S_,
//           tile_dim0=E_, tile_dim1=rows, tensor_dim0_stride=E_
__device__ inline void tdm_load_ktile(unsigned lds_off, const __bf16* gptr,
                                      unsigned rows) {
  unsigned long long ga = (unsigned long long)(size_t)gptr;
  u32x4 g0;
  g0[0] = 1u;                                       // count=1, user descriptor
  g0[1] = lds_off;                                  // LDS byte address
  g0[2] = (unsigned)ga;                             // global_addr[31:0]
  g0[3] = (unsigned)((ga >> 32) & 0x01FFFFFFu)      // global_addr[56:32]
          | (2u << 30);                             // type=2 ("image")
  i32x8 g1;
  g1[0] = (int)(1u << 16);                          // data_size=1 -> 2 bytes
  g1[1] = (int)(((unsigned)E_ & 0xFFFFu) << 16);    // tensor_dim0[15:0]
  g1[2] = (int)((((unsigned)E_ >> 16) & 0xFFFFu)
          | (((unsigned)S_ & 0xFFFFu) << 16));      // dim0 hi | tensor_dim1 lo
  g1[3] = (int)((((unsigned)S_ >> 16) & 0xFFFFu)
          | (((unsigned)E_ & 0xFFFFu) << 16));      // dim1 hi | tile_dim0
  g1[4] = (int)(rows & 0xFFFFu);                    // tile_dim1 (tile_dim2=0)
  g1[5] = (int)E_;                                  // tensor_dim0_stride[31:0]
  g1[6] = 0;                                        // stride0 hi | stride1 lo
  g1[7] = 0;                                        // stride1 hi
  i32x4 z4 = {0, 0, 0, 0};
#if defined(__clang_major__) && (__clang_major__ >= 23)
  i32x8 z8 = {0, 0, 0, 0, 0, 0, 0, 0};
  __builtin_amdgcn_tensor_load_to_lds(g0, g1, z4, z4, z8, 0);
#else
  __builtin_amdgcn_tensor_load_to_lds(g0, g1, z4, z4, 0);
#endif
}

// ======================================================================
// Kernel 1: qa = cos(x @ Wp^T + theta_rx), output bf16.
// grid (ROWS/16, 6), block 256 (8 waves); one wave = one 16x16 tile.
// ======================================================================
__global__ __launch_bounds__(256) void k_proj(const float* __restrict__ x,
                                              const float* __restrict__ Wp,
                                              const float* __restrict__ th,
                                              __bf16* __restrict__ qa) {
  int wave = threadIdx.x >> 5, lane = threadIdx.x & 31;
  int r0 = blockIdx.x * 16;
  int c0 = (blockIdx.y * 8 + wave) * 16;
  f32x8 acc = {};
  for (int k0 = 0; k0 < E_; k0 += 32) {
    bf16x16 a = ld_a_rm_f32(x + (size_t)r0 * E_ + k0, E_, lane);
    bf16x16 b = ld_b_nk_f32(Wp + (size_t)c0 * E_ + k0, E_, lane); // Wp[f,e]
    acc = wmma_bf16(a, b, acc);
  }
  int h = lane >> 4, n = lane & 15;
  int col = c0 + n;
  float t = th[col];
#pragma unroll
  for (int r = 0; r < 8; ++r) {
    int row = r0 + r + 8 * h;
    qa[(size_t)row * E_ + col] = (__bf16)__cosf(acc[r] + t);
  }
}

// ======================================================================
// Kernel 2: flash attention over qa (Q=K=V=qa), fused residual + LN1.
// One wave (block=32) owns 16 query rows; accumulator [16 x 768] in LDS.
// K/V tiles (32 x 768 bf16) are DMA'd into double-buffered LDS by the
// Tensor Data Mover and consumed twice (QK^T and P@V) from LDS.
// ======================================================================
__global__ __launch_bounds__(32) void k_attn(const __bf16* __restrict__ qa,
                                             const float*  __restrict__ x,
                                             const float*  __restrict__ g1,
                                             const float*  __restrict__ b1v,
                                             float*        __restrict__ x1) {
  __shared__ __bf16 kL[2][32 * E_];  // 2 x 48 KB double-buffered K/V tile
  __shared__ float  accL[16 * E_];   // 48 KB running O accumulator
  __shared__ __bf16 pL[16 * 32];     // staged P tile (row-major, ld=32)
  __shared__ float  mL[16], lL[16];

  int lane = threadIdx.x;
  int b   = blockIdx.x >> 7;                 // S/16 = 128 tiles per batch
  int q0  = (blockIdx.x & 127) * 16;
  const __bf16* qaB = qa + (size_t)b * S_ * E_;

  for (int i = lane; i < 16 * E_; i += 32) accL[i] = 0.f;
  if (lane < 16) { mL[lane] = -3.0e30f; lL[lane] = 0.f; }

  // persistent Q fragments
  bf16x16 qf[E_ / 32];
#pragma unroll
  for (int e = 0; e < E_ / 32; ++e)
    qf[e] = ld_a_rm_bf16(qaB + (size_t)q0 * E_ + e * 32, E_, lane);
  __syncthreads();

  int h = lane >> 4, n = lane & 15;
  float alpha[8];

  // prologue: DMA first K tile
  tdm_load_ktile((unsigned)(size_t)(&kL[0][0]), qaB, 32);

  int cur = 0;
  for (int t0 = 0; t0 < S_; t0 += 32, cur ^= 1) {
    if ((t0 + 32) < S_) {
      // DMA next tile into the other buffer while we compute this one,
      // then wait until only that in-flight transfer remains outstanding.
      tdm_load_ktile((unsigned)(size_t)(&kL[cur ^ 1][0]),
                     qaB + (size_t)(t0 + 32) * E_, 32);
      __builtin_amdgcn_s_wait_tensorcnt(1);
    } else {
      __builtin_amdgcn_s_wait_tensorcnt(0);  // last tile: drain the TDM
    }
    __builtin_amdgcn_wave_barrier();
    const __bf16* kT = &kL[cur][0];

    // ---- scores S[16 x 32] = Q * K^T, contraction over E (from LDS) ----
    f32x8 s0 = {}, s1 = {};
    for (int e = 0; e < E_ / 32; ++e) {
      bf16x16 k0f = ld_b_nk_bf16(kT + e * 32, E_, lane);            // t 0..15
      bf16x16 k1f = ld_b_nk_bf16(kT + 16 * E_ + e * 32, E_, lane);  // t 16..31
      s0 = wmma_bf16(qf[e], k0f, s0);
      s1 = wmma_bf16(qf[e], k1f, s1);
    }
    // ---- online softmax update (rows live across 16-lane halves) ----
#pragma unroll
    for (int r = 0; r < 8; ++r) {
      int m = r + 8 * h;
      float v0 = s0[r] * SCALE_, v1 = s1[r] * SCALE_;
      float tmax = red_max16(fmaxf(v0, v1));
      float mOld = mL[m];
      float mNew = fmaxf(mOld, tmax);
      float a = __expf(mOld - mNew);
      float p0 = __expf(v0 - mNew), p1 = __expf(v1 - mNew);
      float rs = red_sum16(p0 + p1);
      if (n == 0) { mL[m] = mNew; lL[m] = lL[m] * a + rs; }
      alpha[r] = a;
      pL[m * 32 + n]      = (__bf16)p0;
      pL[m * 32 + 16 + n] = (__bf16)p1;
    }
    __builtin_amdgcn_wave_barrier();   // order LDS P stage vs. re-read
    // ---- O = alpha*O + P @ V  (V rows come from the same LDS tile) ----
    bf16x16 pf = ld_a_rm_bf16(pL, 32, lane);
    for (int d = 0; d < E_ / 16; ++d) {
      int d0 = d * 16;
      f32x8 c;
#pragma unroll
      for (int r = 0; r < 8; ++r)
        c[r] = accL[(r + 8 * h) * E_ + d0 + n] * alpha[r];
      bf16x16 vf = ld_b_kn_bf16(kT + d0, E_, lane);
      c = wmma_bf16(pf, vf, c);
#pragma unroll
      for (int r = 0; r < 8; ++r)
        accL[(r + 8 * h) * E_ + d0 + n] = c[r];
    }
    // all DS reads of kT must land before TDM may overwrite this buffer
    asm volatile("s_wait_dscnt 0" ::: "memory");
  }
  __syncthreads();

  // ---- finalize: x1 = LayerNorm(x + O/l) ----
  const float* xB  = x  + ((size_t)b * S_ + q0) * E_;
  float*       x1B = x1 + ((size_t)b * S_ + q0) * E_;
  for (int m = 0; m < 16; ++m) {
    float linv = 1.0f / lL[m];
    float s = 0.f;
    for (int e = lane; e < E_; e += 32) {
      float y = xB[(size_t)m * E_ + e] + accL[m * E_ + e] * linv;
      accL[m * E_ + e] = y;
      s += y;
    }
    float mean = red_sum32(s) * (1.0f / E_);
    float var = 0.f;
    for (int e = lane; e < E_; e += 32) {
      float d = accL[m * E_ + e] - mean;
      var += d * d;
    }
    float rstd = rsqrtf(red_sum32(var) * (1.0f / E_) + EPS_);
    for (int e = lane; e < E_; e += 32)
      x1B[(size_t)m * E_ + e] =
          (accL[m * E_ + e] - mean) * rstd * g1[e] + b1v[e];
  }
}

// ======================================================================
// Kernel 3: fused FFN + LN2. Block 256 (8 waves) owns 16 rows.
// qf -> h (16x3072 bf16 in LDS) -> out tile -> residual + LN2.
// ======================================================================
__global__ __launch_bounds__(256) void k_ffn(const float* __restrict__ x1,
                                             const float* __restrict__ thy,
                                             const float* __restrict__ W1,
                                             const float* __restrict__ bb1,
                                             const float* __restrict__ W2,
                                             const float* __restrict__ bb2,
                                             const float* __restrict__ g2,
                                             const float* __restrict__ b2v,
                                             float* __restrict__ out) {
  __shared__ __bf16 qfL[16 * NQ_];   //   4 KB
  __shared__ __bf16 hL[16 * FF_];    //  96 KB (CDNA5: 320 KB/WGP available)
  __shared__ float  yL[16 * E_];     //  48 KB

  int tid = threadIdx.x, wave = tid >> 5, lane = tid & 31;
  int r0 = blockIdx.x * 16;
  const float* xrow = x1 + (size_t)r0 * E_;

  // stage qf = cos(x1[:, :NQ] + theta_ry)  (bf16)
  for (int i = tid; i < 16 * NQ_; i += 256) {
    int m = i >> 7, q = i & (NQ_ - 1);
    qfL[i] = (__bf16)__cosf(xrow[(size_t)m * E_ + q] + thy[q]);
  }
  __syncthreads();

  int h = lane >> 4, n = lane & 15;

  // stage 1: h = relu(qf @ W1^T + b1) -> LDS bf16
  for (int nt = wave; nt < FF_ / 16; nt += 8) {
    int f0 = nt * 16;
    f32x8 acc = {};
#pragma unroll
    for (int k0 = 0; k0 < NQ_; k0 += 32) {
      bf16x16 a = ld_a_rm_bf16(qfL + k0, NQ_, lane);
      bf16x16 b = ld_b_nk_f32(W1 + (size_t)f0 * NQ_ + k0, NQ_, lane); // W1[f,q]
      acc = wmma_bf16(a, b, acc);
    }
    float bias = bb1[f0 + n];
#pragma unroll
    for (int r = 0; r < 8; ++r)
      hL[(r + 8 * h) * FF_ + f0 + n] = (__bf16)fmaxf(acc[r] + bias, 0.f);
  }
  __syncthreads();

  // stage 2: y = h @ W2^T + b2 + x1 -> LDS f32
  for (int nt = wave; nt < E_ / 16; nt += 8) {
    int e0 = nt * 16;
    f32x8 acc = {};
    for (int k0 = 0; k0 < FF_; k0 += 32) {
      // prefetch the upcoming W2 panel (global_prefetch_b8)
      __builtin_prefetch(W2 + (size_t)e0 * FF_ + k0 + 512, 0, 1);
      bf16x16 a = ld_a_rm_bf16(hL + k0, FF_, lane);
      bf16x16 b = ld_b_nk_f32(W2 + (size_t)e0 * FF_ + k0, FF_, lane); // W2[e,f]
      acc = wmma_bf16(a, b, acc);
    }
    float bias = bb2[e0 + n];
#pragma unroll
    for (int r = 0; r < 8; ++r) {
      int m = r + 8 * h;
      yL[m * E_ + e0 + n] = acc[r] + bias + xrow[(size_t)m * E_ + e0 + n];
    }
  }
  __syncthreads();

  // stage 3: LN2, two rows per wave
  for (int m = wave * 2; m < wave * 2 + 2; ++m) {
    float s = 0.f;
    for (int e = lane; e < E_; e += 32) s += yL[m * E_ + e];
    float mean = red_sum32(s) * (1.0f / E_);
    float var = 0.f;
    for (int e = lane; e < E_; e += 32) {
      float d = yL[m * E_ + e] - mean;
      var += d * d;
    }
    float rstd = rsqrtf(red_sum32(var) * (1.0f / E_) + EPS_);
    for (int e = lane; e < E_; e += 32)
      out[((size_t)r0 + m) * E_ + e] =
          (yL[m * E_ + e] - mean) * rstd * g2[e] + b2v[e];
  }
}

// ======================================================================
extern "C" void kernel_launch(void* const* d_in, const int* in_sizes, int n_in,
                              void* d_out, int out_size, void* d_ws, size_t ws_size,
                              hipStream_t stream) {
  const float* x      = (const float*)d_in[0];
  const float* Wp     = (const float*)d_in[1];
  const float* th_rx  = (const float*)d_in[2];
  const float* th_ry  = (const float*)d_in[3];
  const float* W1     = (const float*)d_in[4];
  const float* b1     = (const float*)d_in[5];
  const float* W2     = (const float*)d_in[6];
  const float* b2     = (const float*)d_in[7];
  const float* ln1_g  = (const float*)d_in[8];
  const float* ln1_b  = (const float*)d_in[9];
  const float* ln2_g  = (const float*)d_in[10];
  const float* ln2_b  = (const float*)d_in[11];
  float* out = (float*)d_out;

  // workspace layout: qa (bf16, 24 MB) | x1 (f32, 48 MB)
  __bf16* qa = (__bf16*)d_ws;
  float*  x1 = (float*)((char*)d_ws + (size_t)32 * 1024 * 1024);

  k_proj<<<dim3(ROWS_ / 16, 6), 256, 0, stream>>>(x, Wp, th_rx, qa);
  k_attn<<<ROWS_ / 16, 32, 0, stream>>>(qa, x, ln1_g, ln1_b, x1);
  k_ffn <<<ROWS_ / 16, 256, 0, stream>>>(x1, th_ry, W1, b1, W2, b2,
                                         ln2_g, ln2_b, out);
}
